// KeypointNetwithIOLoss_13889924235294
// MI455X (gfx1250) — matrix operand order repaired and checked by reading, not compiled, not run on hardware
//
#include <hip/hip_runtime.h>
#include <math.h>

#define BB   8
#define HCW  40          // Hc = Wc = 40
#define NF   1444        // (40-2)*(40-2)
#define NP   1456        // NF padded to 91*16
#define NT   91          // 16-wide tiles
#define CC   64
#define EPSF 1e-8f
#define IMGS 159.5f      // (320-1)/2

typedef __attribute__((ext_vector_type(16))) _Float16 v16h;
typedef __attribute__((ext_vector_type(8)))  float    v8f;

// ---------------- init: zero accumulators + f16 pad rows ----------------
__global__ void k_init(float* acc, _Float16* refh, _Float16* tarh) {
    int t = blockIdx.x * blockDim.x + threadIdx.x;
    if (t < 4) acc[t] = 0.0f;
    // pad rows NF..NP-1 for both f16 arrays: 2 * B * 12 * 64 elements
    const int per = BB * (NP - NF) * CC;   // 6144
    if (t < 2 * per) {
        int arr = t / per, r = t % per;
        int b = r / ((NP - NF) * CC);
        int rem = r % ((NP - NF) * CC);
        int row = NF + rem / CC, c = rem % CC;
        size_t off = ((size_t)b * NP + row) * CC + c;
        (arr ? tarh : refh)[off] = (_Float16)0.0f;
    }
}

// ---------------- homography warp of all 1600 source points ----------------
__global__ void k_warp(const float* __restrict__ src_uv,
                       const float* __restrict__ hom,
                       float* __restrict__ warped,       // (B,1600,2) pixel coords
                       float* __restrict__ warped_norm,  // (B,1600,2) normalized
                       float* __restrict__ tar_pts)      // (B,NF,2) interior warped
{
    int idx = blockIdx.x * blockDim.x + threadIdx.x;
    if (idx >= BB * HCW * HCW) return;
    int b = idx / (HCW * HCW), i = idx % (HCW * HCW);
    int y = i / HCW, x = i % HCW;
    float u = src_uv[(size_t)idx * 2 + 0];
    float v = src_uv[(size_t)idx * 2 + 1];
    float nx = u * (1.0f / IMGS) - 1.0f;
    float ny = v * (1.0f / IMGS) - 1.0f;
    const float* h = hom + (size_t)b * 9;
    float w0 = h[0] * nx + h[1] * ny + h[2];
    float w1 = h[3] * nx + h[4] * ny + h[5];
    float w2 = h[6] * nx + h[7] * ny + h[8];
    float inv = 1.0f / (w2 + EPSF);
    float wnx = w0 * inv, wny = w1 * inv;
    warped_norm[(size_t)idx * 2 + 0] = wnx;
    warped_norm[(size_t)idx * 2 + 1] = wny;
    float px = (wnx + 1.0f) * IMGS, py = (wny + 1.0f) * IMGS;
    warped[(size_t)idx * 2 + 0] = px;
    warped[(size_t)idx * 2 + 1] = py;
    if (y >= 1 && y <= HCW - 2 && x >= 1 && x <= HCW - 2) {
        int fi = (y - 1) * (HCW - 2) + (x - 1);
        tar_pts[((size_t)b * NF + fi) * 2 + 0] = px;
        tar_pts[((size_t)b * NF + fi) * 2 + 1] = py;
    }
}

// ---------------- loc loss: per-row min over 1600 target points ----------------
// min_j sqrt(dx^2+dy^2) == sqrt(min_j (dx^2+dy^2)) : sqrt hoisted out of loop
__global__ void k_loc(const float* __restrict__ warped,
                      const float* __restrict__ tgt_uv,
                      float* acc)
{
    __shared__ float tsx[HCW * HCW];
    __shared__ float tsy[HCW * HCW];
    int b = blockIdx.y;
    for (int j = threadIdx.x; j < HCW * HCW; j += blockDim.x) {
        tsx[j] = tgt_uv[((size_t)b * HCW * HCW + j) * 2 + 0];
        tsy[j] = tgt_uv[((size_t)b * HCW * HCW + j) * 2 + 1];
    }
    __syncthreads();
    int i = blockIdx.x * blockDim.x + threadIdx.x;
    if (i >= HCW * HCW) return;
    float sx = warped[((size_t)b * HCW * HCW + i) * 2 + 0];
    float sy = warped[((size_t)b * HCW * HCW + i) * 2 + 1];
    float m2 = INFINITY;
    for (int j = 0; j < HCW * HCW; ++j) {
        float dx = fabsf(sx - tsx[j]) + EPSF;
        float dy = fabsf(sy - tsy[j]) + EPSF;
        m2 = fminf(m2, dx * dx + dy * dy);
    }
    float dmin = sqrtf(m2);
    int yy = i / HCW, xx = i % HCW;
    bool border = (yy >= 1 && yy <= HCW - 2 && xx >= 1 && xx <= HCW - 2);
    if (border && dmin < 4.0f) {
        atomicAdd(&acc[0], dmin);
        atomicAdd(&acc[1], 1.0f);
    }
}

// ---------------- bilinear sample (2 channels per lane) ----------------
__device__ inline void bilerp2(const float* __restrict__ feat, int b,
                               float gx, float gy, int lane,
                               float& o0, float& o1)
{
    float fx = fminf(fmaxf((gx + 1.0f) * 19.5f, 0.0f), 39.0f);
    float fy = fminf(fmaxf((gy + 1.0f) * 19.5f, 0.0f), 39.0f);
    int x0 = (int)floorf(fx), y0 = (int)floorf(fy);
    int x1 = min(x0 + 1, HCW - 1), y1 = min(y0 + 1, HCW - 1);
    float wx = fx - (float)x0, wy = fy - (float)y0;
    const float* p00 = feat + ((size_t)(b * HCW + y0) * HCW + x0) * CC;
    const float* p01 = feat + ((size_t)(b * HCW + y0) * HCW + x1) * CC;
    const float* p10 = feat + ((size_t)(b * HCW + y1) * HCW + x0) * CC;
    const float* p11 = feat + ((size_t)(b * HCW + y1) * HCW + x1) * CC;
    float a = (1.0f - wx) * (1.0f - wy), bw = wx * (1.0f - wy);
    float cw = (1.0f - wx) * wy, dw = wx * wy;
    int c0 = lane, c1 = lane + 32;
    o0 = p00[c0] * a + p01[c0] * bw + p10[c0] * cw + p11[c0] * dw;
    o1 = p00[c1] * a + p01[c1] * bw + p10[c1] * cw + p11[c1] * dw;
}

// ---------------- grid sample + L2 normalize, wave per point ----------------
__global__ void k_sample(const float* __restrict__ src_feat,
                         const float* __restrict__ tgt_feat,
                         const float* __restrict__ src_uv,
                         const float* __restrict__ warped_norm,
                         float* __restrict__ ref_f32,
                         float* __restrict__ tar_f32,
                         _Float16* __restrict__ refh,
                         _Float16* __restrict__ tarh)
{
    int lane = threadIdx.x & 31;
    int w = (blockIdx.x * blockDim.x + threadIdx.x) >> 5;
    if (w >= BB * NF) return;
    int b = w / NF, i = w % NF;
    int y = i / (HCW - 2) + 1, x = i % (HCW - 2) + 1;
    int gi = (b * HCW + y) * HCW + x;
    float u = src_uv[(size_t)gi * 2 + 0];
    float v = src_uv[(size_t)gi * 2 + 1];
    float gx = u * (1.0f / IMGS) - 1.0f;
    float gy = v * (1.0f / IMGS) - 1.0f;
    float r0, r1, t0, t1;
    bilerp2(src_feat, b, gx, gy, lane, r0, r1);
    float wnx = warped_norm[(size_t)gi * 2 + 0];
    float wny = warped_norm[(size_t)gi * 2 + 1];
    bilerp2(tgt_feat, b, wnx, wny, lane, t0, t1);
    // _l2n: v += EPS; v = v/||v|| + EPS
    r0 += EPSF; r1 += EPSF; t0 += EPSF; t1 += EPSF;
    float sr = r0 * r0 + r1 * r1, st = t0 * t0 + t1 * t1;
    #pragma unroll
    for (int m = 1; m < 32; m <<= 1) {
        sr += __shfl_xor(sr, m, 32);
        st += __shfl_xor(st, m, 32);
    }
    float irn = 1.0f / sqrtf(sr), itn = 1.0f / sqrtf(st);
    r0 = r0 * irn + EPSF; r1 = r1 * irn + EPSF;
    t0 = t0 * itn + EPSF; t1 = t1 * itn + EPSF;
    size_t rowf = ((size_t)b * NF + i) * CC;
    size_t rowh = ((size_t)b * NP + i) * CC;
    ref_f32[rowf + lane] = r0;  ref_f32[rowf + lane + 32] = r1;
    tar_f32[rowf + lane] = t0;  tar_f32[rowf + lane + 32] = t1;
    refh[rowh + lane] = (_Float16)r0;  refh[rowh + lane + 32] = (_Float16)r1;
    tarh[rowh + lane] = (_Float16)t0;  tarh[rowh + lane + 32] = (_Float16)t1;
}

// ---------------- fused WMMA GEMM + hardest-negative mining ----------------
// one wave per (batch, 16-row tile); loops over 91 column tiles.
// Min/argmin done in squared-distance space (monotone in sqrt), sentinel 2.0 -> 4.0.
__global__ void k_gemm_neg(const _Float16* __restrict__ refh,
                           const _Float16* __restrict__ tarh,
                           const float* __restrict__ tar_pts,
                           int* __restrict__ neg_idx)
{
    int lane = threadIdx.x & 31;
    int w = blockIdx.x * (blockDim.x >> 5) + (threadIdx.x >> 5);   // 0..727
    int b = w / NT, tile = w % NT;
    int half = lane >> 4;          // 0: rows v / K 0-7,16-23, 1: rows v+8 / K 8-15,24-31 (per 32-K step)
    int l15 = lane & 15;
    int base8 = half * 8;

    // A fragments: row = tile*16 + l15; elems 0-7 -> K base8.., 8-15 -> K 16+base8..
    int arow = tile * 16 + l15;
    const _Float16* ar = refh + ((size_t)b * NP + arow) * CC;
    v16h a0, a1;
    #pragma unroll
    for (int e = 0; e < 8; ++e) {
        a0[e]     = ar[base8 + e];
        a0[8 + e] = ar[16 + base8 + e];
        a1[e]     = ar[32 + base8 + e];
        a1[8 + e] = ar[48 + base8 + e];
    }
    // row keypoints (shared within each 16-lane half)
    float tx[8], ty[8];
    #pragma unroll
    for (int v = 0; v < 8; ++v) {
        int ri = tile * 16 + v + 8 * half;
        bool ok = ri < NF;
        tx[v] = ok ? tar_pts[((size_t)b * NF + ri) * 2 + 0] : 0.0f;
        ty[v] = ok ? tar_pts[((size_t)b * NF + ri) * 2 + 1] : 0.0f;
    }
    float minv[8]; int mini[8];
    #pragma unroll
    for (int v = 0; v < 8; ++v) { minv[v] = INFINITY; mini[v] = 0; }

    int kb = half * 16;   // B fragment: col = l15, elems 0-7 -> K base8.., 8-15 -> K 16+base8..
    for (int jt = 0; jt < NT; ++jt) {
        int col = jt * 16 + l15;
        const _Float16* br = tarh + ((size_t)b * NP + col) * CC;
        if (jt + 1 < NT)
            __builtin_prefetch(tarh + ((size_t)b * NP + (jt + 1) * 16 + l15) * CC, 0, 0);
        v16h b0, b1;
        #pragma unroll
        for (int e = 0; e < 8; ++e) {
            b0[e]     = br[base8 + e];
            b0[8 + e] = br[16 + base8 + e];
            b1[e]     = br[32 + base8 + e];
            b1[8 + e] = br[48 + base8 + e];
        }
        v8f c = {};
        c = __builtin_amdgcn_wmma_f32_16x16x32_f16(false, a0, false, b0,
                                                   (short)0, c, false, false);
        c = __builtin_amdgcn_wmma_f32_16x16x32_f16(false, a1, false, b1,
                                                   (short)0, c, false, false);
        bool jok = col < NF;
        float mx = jok ? tar_pts[((size_t)b * NF + col) * 2 + 0] : 1e30f;
        float my = jok ? tar_pts[((size_t)b * NF + col) * 2 + 1] : 1e30f;
        #pragma unroll
        for (int v = 0; v < 8; ++v) {
            float d = fminf(fmaxf(c[v], -1.0f), 1.0f);
            float dist2 = 2.0f - 2.0f * d + EPSF;              // squared distance
            bool correct = (fabsf(mx - tx[v]) <= 4.0f) && (fabsf(my - ty[v]) <= 4.0f);
            float val = !jok ? INFINITY : (correct ? 4.0f : dist2);
            if (val < minv[v] || (val == minv[v] && col < mini[v])) {
                minv[v] = val; mini[v] = col;
            }
        }
    }
    // butterfly min within each 16-lane half (covers all 16 columns per tile slot)
    #pragma unroll
    for (int m = 1; m < 16; m <<= 1) {
        #pragma unroll
        for (int v = 0; v < 8; ++v) {
            float ov = __shfl_xor(minv[v], m, 32);
            int   oi = __shfl_xor(mini[v], m, 32);
            if (ov < minv[v] || (ov == minv[v] && oi < mini[v])) {
                minv[v] = ov; mini[v] = oi;
            }
        }
    }
    if (l15 == 0) {
        #pragma unroll
        for (int v = 0; v < 8; ++v) {
            int ri = tile * 16 + v + 8 * half;
            if (ri < NF) neg_idx[b * NF + ri] = mini[v];
        }
    }
}

// ---------------- metric (triplet hinge) loss, wave per row ----------------
__global__ void k_metric(const float* __restrict__ ref_f32,
                         const float* __restrict__ tar_f32,
                         const int* __restrict__ neg_idx,
                         float* acc)
{
    int lane = threadIdx.x & 31;
    int w = blockIdx.x * (blockDim.x >> 5) + (threadIdx.x >> 5);
    if (w >= BB * NF) return;
    int b = w / NF, i = w % NF;
    const float* r = ref_f32 + ((size_t)b * NF + i) * CC;
    const float* t = tar_f32 + ((size_t)b * NF + i) * CC;
    int ni = neg_idx[w];
    const float* n = tar_f32 + ((size_t)b * NF + ni) * CC;
    float dp = 0.0f, dn = 0.0f;
    #pragma unroll
    for (int c = 0; c < 2; ++c) {
        int ch = lane + 32 * c;
        float a = r[ch] - t[ch]; dp += a * a;
        float g = r[ch] - n[ch]; dn += g * g;
    }
    #pragma unroll
    for (int m = 1; m < 32; m <<= 1) {
        dp += __shfl_xor(dp, m, 32);
        dn += __shfl_xor(dn, m, 32);
    }
    if (lane == 0) {
        float h = fmaxf(0.0f, 0.2f + sqrtf(dp) - sqrtf(dn));
        atomicAdd(&acc[2], h);
    }
}

// ---------------- final combine ----------------
__global__ void k_final(const float* acc, float* out) {
    // total = loc_sum / max(cnt,1) + DESC_W * metric_mean * 2
    float loc = acc[0] / fmaxf(acc[1], 1.0f);
    float met = acc[2] / (float)(BB * NF);
    out[0] = loc + 4.0f * met;
}

extern "C" void kernel_launch(void* const* d_in, const int* in_sizes, int n_in,
                              void* d_out, int out_size, void* d_ws, size_t ws_size,
                              hipStream_t stream) {
    const float* src_feat = (const float*)d_in[0];
    const float* tgt_feat = (const float*)d_in[1];
    const float* src_uv   = (const float*)d_in[2];
    const float* tgt_uv   = (const float*)d_in[3];
    const float* hom      = (const float*)d_in[4];
    float* out = (float*)d_out;

    char* p = (char*)d_ws;
    auto carve = [&](size_t bytes) -> void* {
        void* r = (void*)p;
        p += (bytes + 255) & ~(size_t)255;
        return r;
    };
    float* warped      = (float*)carve((size_t)BB * HCW * HCW * 2 * 4);
    float* warped_norm = (float*)carve((size_t)BB * HCW * HCW * 2 * 4);
    float* tar_pts     = (float*)carve((size_t)BB * NF * 2 * 4);
    float* ref_f32     = (float*)carve((size_t)BB * NF * CC * 4);
    float* tar_f32     = (float*)carve((size_t)BB * NF * CC * 4);
    _Float16* refh     = (_Float16*)carve((size_t)BB * NP * CC * 2);
    _Float16* tarh     = (_Float16*)carve((size_t)BB * NP * CC * 2);
    int* neg_idx       = (int*)carve((size_t)BB * NF * 4);
    float* acc         = (float*)carve(4 * 4);

    k_init<<<48, 256, 0, stream>>>(acc, refh, tarh);
    k_warp<<<(BB * HCW * HCW + 255) / 256, 256, 0, stream>>>(
        src_uv, hom, warped, warped_norm, tar_pts);
    k_loc<<<dim3((HCW * HCW + 255) / 256, BB), 256, 0, stream>>>(
        warped, tgt_uv, acc);
    k_sample<<<(BB * NF * 32 + 255) / 256, 256, 0, stream>>>(
        src_feat, tgt_feat, src_uv, warped_norm, ref_f32, tar_f32, refh, tarh);
    k_gemm_neg<<<(BB * NT) / 8, 256, 0, stream>>>(refh, tarh, tar_pts, neg_idx);
    k_metric<<<(BB * NF) / 8, 256, 0, stream>>>(ref_f32, tar_f32, neg_idx, acc);
    k_final<<<1, 1, 0, stream>>>(acc, out);
}
